// ImprovedMPNN_80187039416388
// MI455X (gfx1250) — matrix operand order repaired and checked by reading, not compiled, not run on hardware
//
#include <hip/hip_runtime.h>
#include <hip/hip_bf16.h>

typedef __attribute__((ext_vector_type(16))) _Float16 v16h;
typedef __attribute__((ext_vector_type(8)))  _Float16 v8h;
typedef __attribute__((ext_vector_type(8)))  float    v8f;

#define WMMA(a,b,c) __builtin_amdgcn_wmma_f32_16x16x32_f16(false,(a),false,(b),(short)0,(c),false,false)
#define LDS_FENCE() asm volatile("s_wait_dscnt 0x0" ::: "memory")

__device__ __forceinline__ float redsum16(float v){
  v += __shfl_xor(v,1); v += __shfl_xor(v,2);
  v += __shfl_xor(v,4); v += __shfl_xor(v,8);
  return v;
}
__device__ __forceinline__ float redmax16(float v){
  v = fmaxf(v,__shfl_xor(v,1)); v = fmaxf(v,__shfl_xor(v,2));
  v = fmaxf(v,__shfl_xor(v,4)); v = fmaxf(v,__shfl_xor(v,8));
  return v;
}
__device__ __forceinline__ float redsum32(float v){
  v += __shfl_xor(v,1); v += __shfl_xor(v,2);
  v += __shfl_xor(v,4); v += __shfl_xor(v,8); v += __shfl_xor(v,16);
  return v;
}
__device__ __forceinline__ float sigm(float x){ return 1.f/(1.f+__expf(-x)); }

// ---- fragment loaders (all pointers 16B aligned by construction; no branches) ----
// A-frag (16x32 f16): lane holds row, element e -> k = 16*hi + e
__device__ __forceinline__ v16h a_frag_g(const float* row_k0, int hi, float m){
  const float* p0 = row_k0 + 16*hi;
  const float* p1 = row_k0 + 16*hi + 8;
  float4 x0 = ((const float4*)p0)[0], x1 = ((const float4*)p0)[1];
  float4 x2 = ((const float4*)p1)[0], x3 = ((const float4*)p1)[1];
  v16h a;
  a[0]=(_Float16)(x0.x*m); a[1]=(_Float16)(x0.y*m); a[2]=(_Float16)(x0.z*m); a[3]=(_Float16)(x0.w*m);
  a[4]=(_Float16)(x1.x*m); a[5]=(_Float16)(x1.y*m); a[6]=(_Float16)(x1.z*m); a[7]=(_Float16)(x1.w*m);
  a[8]=(_Float16)(x2.x*m); a[9]=(_Float16)(x2.y*m); a[10]=(_Float16)(x2.z*m); a[11]=(_Float16)(x2.w*m);
  a[12]=(_Float16)(x3.x*m); a[13]=(_Float16)(x3.y*m); a[14]=(_Float16)(x3.z*m); a[15]=(_Float16)(x3.w*m);
  return a;
}
// 16 consecutive f32 -> f16 frag (B-frag: caller passes row + k0 + 16*hi)
__device__ __forceinline__ v16h cvt16(const float* p, float m){
  float4 x0 = ((const float4*)p)[0], x1 = ((const float4*)p)[1];
  float4 x2 = ((const float4*)p)[2], x3 = ((const float4*)p)[3];
  v16h b;
  b[0]=(_Float16)(x0.x*m); b[1]=(_Float16)(x0.y*m); b[2]=(_Float16)(x0.z*m); b[3]=(_Float16)(x0.w*m);
  b[4]=(_Float16)(x1.x*m); b[5]=(_Float16)(x1.y*m); b[6]=(_Float16)(x1.z*m); b[7]=(_Float16)(x1.w*m);
  b[8]=(_Float16)(x2.x*m); b[9]=(_Float16)(x2.y*m); b[10]=(_Float16)(x2.z*m); b[11]=(_Float16)(x2.w*m);
  b[12]=(_Float16)(x3.x*m); b[13]=(_Float16)(x3.y*m); b[14]=(_Float16)(x3.z*m); b[15]=(_Float16)(x3.w*m);
  return b;
}
// A-frag from f16 LDS row (row + k0), 16B vector LDS loads
__device__ __forceinline__ v16h a_frag_ldsh(const _Float16* row_k0, int hi){
  v8h a0 = *(const v8h*)(row_k0 + 16*hi);
  v8h a1 = *(const v8h*)(row_k0 + 16*hi + 8);
  v16h a;
#pragma unroll
  for (int e = 0; e < 8; ++e){ a[e] = a0[e]; a[8+e] = a1[e]; }
  return a;
}

// ---------------------------------------------------------------- zero
__global__ void k_zero(float* __restrict__ p, int n){
  int i = blockIdx.x*blockDim.x + threadIdx.x;
  if (i < n) p[i] = 0.f;
}

// ---------------------------------------------------------------- generic GEMM: C = A[M,K] @ W[N,K]^T + bias (+C)
// requires K % 32 == 0, M % 16 == 0, N % 32 == 0, bias != null; 2 N-tiles per wave
template<bool ACC>
__global__ void __launch_bounds__(128)
k_gemm(const float* __restrict__ A, const float* __restrict__ W,
       const float* __restrict__ bias, float* __restrict__ C,
       int M, int N, int K)
{
  int w    = threadIdx.x >> 5;
  int lane = threadIdx.x & 31;
  int rlo  = lane & 15;
  int hi   = lane >> 4;
  int ntp  = N >> 5;                      // pairs of 16-col tiles
  int tile = blockIdx.x*4 + w;
  if (tile >= (M>>4)*ntp) return;
  int mt = tile / ntp, np = tile % ntp;
  int row = mt*16 + rlo;
  int nc0 = np*32 + rlo;
  int nc1 = nc0 + 16;
  const float* Arow = A + row*K;
  const float* W0   = W + nc0*K;
  const float* W1r  = W + nc1*K;
  v8f acc0 = {}, acc1 = {};
  for (int k0 = 0; k0 < K; k0 += 32){
    v16h a  = a_frag_g(Arow + k0, hi, 1.f);
    v16h b0 = cvt16(W0  + k0 + 16*hi, 1.f);
    v16h b1 = cvt16(W1r + k0 + 16*hi, 1.f);
    acc0 = WMMA(a, b0, acc0);
    acc1 = WMMA(a, b1, acc1);
  }
  float bv0 = bias[nc0], bv1 = bias[nc1];
#pragma unroll
  for (int r = 0; r < 8; ++r){
    int rr   = mt*16 + r + 8*hi;
    float v0 = acc0[r] + bv0;
    float v1 = acc1[r] + bv1;
    if (ACC){ v0 += C[rr*N + nc0]; v1 += C[rr*N + nc1]; }
    C[rr*N + nc0] = v0;
    C[rr*N + nc1] = v1;
  }
}

// ---------------------------------------------------------------- LayerNorm (+residual, +base add, +relu, row-broadcast)
template<bool HASR, bool HASB, bool RELU>
__global__ void __launch_bounds__(256)
k_ln(const float* __restrict__ X, const float* __restrict__ R,
     const float* __restrict__ gam, const float* __restrict__ bet,
     const float* __restrict__ base, float* __restrict__ out,
     int rows, int D, int shift)
{
  int w = blockIdx.x*8 + (threadIdx.x >> 5);
  if (w >= rows) return;
  int lane = threadIdx.x & 31;
  int srow = w >> shift;
  int nper = D >> 5;            // D in {64,128}
  float v[8];
  float s = 0.f;
  for (int i = 0; i < nper; ++i){
    int c = lane + (i<<5);
    float t = X[srow*D + c];
    if (HASR) t += R[w*D + c];
    v[i] = t; s += t;
  }
  s = redsum32(s);
  float mu = s / (float)D;
  float q = 0.f;
  for (int i = 0; i < nper; ++i){ float d = v[i]-mu; q += d*d; }
  q = redsum32(q);
  float rs = rsqrtf(q/(float)D + 1e-5f);
  for (int i = 0; i < nper; ++i){
    int c = lane + (i<<5);
    float y = (v[i]-mu)*rs*gam[c] + bet[c];
    if (RELU) y = fmaxf(y, 0.f);
    if (HASB) y += base[w*D + c];
    out[w*D + c] = y;
  }
}

// ---------------------------------------------------------------- fused edge MLP + atomic scatter (segment_sum)
__device__ __forceinline__ void edge_ln_store(v8f* acc, const float* bias,
    const float* g, const float* be, _Float16 (*lds)[136], int rlo, int hi)
{
  float bv[8], gv[8], bev[8];
#pragma unroll
  for (int nt = 0; nt < 8; ++nt){
    int c = nt*16 + rlo;
    bv[nt] = bias[c]; gv[nt] = g[c]; bev[nt] = be[c];
  }
#pragma unroll
  for (int nt = 0; nt < 8; ++nt)
#pragma unroll
    for (int r = 0; r < 8; ++r) acc[nt][r] += bv[nt];
#pragma unroll
  for (int r = 0; r < 8; ++r){
    float s = 0.f;
#pragma unroll
    for (int nt = 0; nt < 8; ++nt) s += acc[nt][r];
    s = redsum16(s);
    float mu = s * (1.f/128.f);
    float q = 0.f;
#pragma unroll
    for (int nt = 0; nt < 8; ++nt){ float d = acc[nt][r]-mu; q += d*d; }
    q = redsum16(q);
    float rs = rsqrtf(q*(1.f/128.f) + 1e-5f);
#pragma unroll
    for (int nt = 0; nt < 8; ++nt){
      float y = (acc[nt][r]-mu)*rs*gv[nt] + bev[nt];
      y = fmaxf(y, 0.f);
      lds[r + 8*hi][nt*16 + rlo] = (_Float16)y;
    }
  }
}

__global__ void __launch_bounds__(128)
k_edge(const float* __restrict__ z, const int* __restrict__ ei, const float* __restrict__ ea,
       const float* __restrict__ W1, const float* __restrict__ b1,
       const float* __restrict__ g1, const float* __restrict__ be1,
       const float* __restrict__ W2, const float* __restrict__ b2,
       const float* __restrict__ g2, const float* __restrict__ be2,
       const float* __restrict__ W3, const float* __restrict__ b3,
       float* __restrict__ agg)
{
  const int E = 65536;
  __shared__ __align__(16) _Float16 hlds[4][16][136];
  int w    = threadIdx.x >> 5;
  int lane = threadIdx.x & 31;
  int rlo  = lane & 15;
  int hi   = lane >> 4;
  int e0   = (blockIdx.x*4 + w)*16;
  int myE  = e0 + rlo;
  int srcL = ei[myE];
  int dstL = ei[E + myE];

  v8f zf = {};
  // ----- layer 1: [z[dst](128) | z[src](128) | ea(16) pad to 288] @ W1^T
  v8f acc[8];
#pragma unroll
  for (int nt = 0; nt < 8; ++nt) acc[nt] = zf;
#pragma unroll
  for (int k0 = 0; k0 < 256; k0 += 32){
    const float* base = (k0 < 128) ? (z + dstL*128 + k0) : (z + srcL*128 + (k0-128));
    v16h a = a_frag_g(base, hi, 1.f);
#pragma unroll
    for (int nt = 0; nt < 8; ++nt){
      v16h b = cvt16(W1 + (nt*16+rlo)*272 + k0 + 16*hi, 1.f);
      acc[nt] = WMMA(a, b, acc[nt]);
    }
  }
  { // tail K-step k0=256: A cols 256..271 = ea, 272..287 = 0; W1 cols valid only for hi==0
    float m = hi ? 0.f : 1.f;                // zero invalid B halves (clamped in-bounds load)
    v16h a = cvt16(ea + myE*16, m);
#pragma unroll
    for (int nt = 0; nt < 8; ++nt){
      v16h b = cvt16(W1 + (nt*16+rlo)*272 + (hi ? 240 : 256), m);
      acc[nt] = WMMA(a, b, acc[nt]);
    }
  }
  edge_ln_store(acc, b1, g1, be1, hlds[w], rlo, hi);
  LDS_FENCE();

  // ----- layer 2: h1 @ W2^T (K=128, A from LDS f16)
  v8f acc2[8];
#pragma unroll
  for (int nt = 0; nt < 8; ++nt) acc2[nt] = zf;
#pragma unroll
  for (int k0 = 0; k0 < 128; k0 += 32){
    v16h a = a_frag_ldsh(&hlds[w][rlo][k0], hi);
#pragma unroll
    for (int nt = 0; nt < 8; ++nt){
      v16h b = cvt16(W2 + (nt*16+rlo)*128 + k0 + 16*hi, 1.f);
      acc2[nt] = WMMA(a, b, acc2[nt]);
    }
  }
  LDS_FENCE();
  edge_ln_store(acc2, b2, g2, be2, hlds[w], rlo, hi);
  LDS_FENCE();

  // ----- layer 3: h2 @ W3^T, atomic scatter to agg[dst]
  v8f acc3[8];
#pragma unroll
  for (int nt = 0; nt < 8; ++nt) acc3[nt] = zf;
#pragma unroll
  for (int k0 = 0; k0 < 128; k0 += 32){
    v16h a = a_frag_ldsh(&hlds[w][rlo][k0], hi);
#pragma unroll
    for (int nt = 0; nt < 8; ++nt){
      v16h b = cvt16(W3 + (nt*16+rlo)*128 + k0 + 16*hi, 1.f);
      acc3[nt] = WMMA(a, b, acc3[nt]);
    }
  }
  float b3v[8];
#pragma unroll
  for (int nt = 0; nt < 8; ++nt) b3v[nt] = b3[nt*16 + rlo];
#pragma unroll
  for (int r = 0; r < 8; ++r){
    int dr = __shfl(dstL, r + 8*hi);
#pragma unroll
    for (int nt = 0; nt < 8; ++nt)
      atomicAdd(&agg[dr*128 + nt*16 + rlo], acc3[nt][r] + b3v[nt]);
  }
}

// ---------------------------------------------------------------- V transpose: vt[(head*32+dh)*4096 + n] = V
__global__ void k_vt(const float* __restrict__ qkv, float* __restrict__ vt){
  int i = blockIdx.x*blockDim.x + threadIdx.x;
  if (i >= 128*4096) return;
  int n = i & 4095;
  int c = i >> 12;
  vt[i] = qkv[n*384 + 256 + c];
}

// ---------------------------------------------------------------- flash attention (4 heads, DH=32, seq=4096)
__global__ void __launch_bounds__(128)
k_attn(const float* __restrict__ qkv, const float* __restrict__ vt,
       float* __restrict__ o)
{
  const int QKV = 384;
  __shared__ __align__(16) _Float16 plds[4][16][40];
  int w    = threadIdx.x >> 5;
  int lane = threadIdx.x & 31;
  int rlo  = lane & 15;
  int hi   = lane >> 4;
  int qt   = blockIdx.x*4 + w;       // 0..255
  int head = blockIdx.y;
  int q0   = qt*16;
  const float scale = 0.17677669529663687f;   // 1/sqrt(32)

  v16h aq = a_frag_g(qkv + (q0+rlo)*QKV + head*32, hi, scale);
  const float* vrow0 = vt + (head*32 + rlo)*4096;
  const float* vrow1 = vt + (head*32 + 16 + rlo)*4096;

  v8f acc0 = {}, acc1 = {};
  float mrow[8], lrow[8];
#pragma unroll
  for (int r = 0; r < 8; ++r){ mrow[r] = -1e30f; lrow[r] = 0.f; }

  for (int kt = 0; kt < 4096; kt += 32){
    v16h b0 = cvt16(qkv + (kt      + rlo)*QKV + 128 + head*32 + 16*hi, 1.f);
    v16h b1 = cvt16(qkv + (kt + 16 + rlo)*QKV + 128 + head*32 + 16*hi, 1.f);
    v8f s0 = {}, s1 = {};
    s0 = WMMA(aq, b0, s0);
    s1 = WMMA(aq, b1, s1);
#pragma unroll
    for (int r = 0; r < 8; ++r){
      float mx   = redmax16(fmaxf(s0[r], s1[r]));
      float mnew = fmaxf(mrow[r], mx);
      float corr = __expf(mrow[r] - mnew);
      float p0   = __expf(s0[r] - mnew);
      float p1   = __expf(s1[r] - mnew);
      float ps   = redsum16(p0 + p1);
      lrow[r] = lrow[r]*corr + ps;
      acc0[r] *= corr; acc1[r] *= corr;
      mrow[r] = mnew;
      plds[w][r + 8*hi][rlo]      = (_Float16)p0;
      plds[w][r + 8*hi][rlo + 16] = (_Float16)p1;
    }
    LDS_FENCE();
    v16h ap = a_frag_ldsh(plds[w][rlo], hi);
    v16h v0 = cvt16(vrow0 + kt + 16*hi, 1.f);
    v16h v1 = cvt16(vrow1 + kt + 16*hi, 1.f);
    acc0 = WMMA(ap, v0, acc0);
    acc1 = WMMA(ap, v1, acc1);
    LDS_FENCE();
  }
#pragma unroll
  for (int r = 0; r < 8; ++r){
    float inv = 1.f / lrow[r];
    int row = q0 + r + 8*hi;
    o[row*128 + head*32 + rlo]      = acc0[r]*inv;
    o[row*128 + head*32 + 16 + rlo] = acc1[r]*inv;
  }
}

// ---------------------------------------------------------------- persistent GRU layer: 64 steps, h in LDS, one block
__global__ void __launch_bounds__(256)
k_gru(const float* __restrict__ zB, const float* __restrict__ Whh,
      const float* __restrict__ Wih, const float* __restrict__ bhh,
      const float* __restrict__ bih, float* __restrict__ hbuf)
{
  __shared__ __align__(16) float hl[64][128];
  int tid = threadIdx.x;
  for (int i = tid; i < 64*128; i += 256) hl[i>>7][i&127] = hbuf[i];
  __syncthreads();
  int w = tid >> 5, lane = tid & 31, rlo = lane & 15, hi = lane >> 4;
  int rt = w >> 1;
  int jb = (w & 1)*4;

  for (int t = 0; t < 64; ++t){
    float hp[4][8];
#pragma unroll
    for (int i = 0; i < 4; ++i){
      int jt = jb + i;
      auto gateacc = [&](const float* W, int gsel, bool fromH) -> v8f {
        v8f acc = {};
#pragma unroll
        for (int k0 = 0; k0 < 128; k0 += 32){
          v16h a = fromH ? a_frag_g(&hl[rt*16 + rlo][k0], hi, 1.f)
                         : a_frag_g(zB + ((rt*16 + rlo)*64 + t)*128 + k0, hi, 1.f);
          v16h b = cvt16(W + (gsel*128 + jt*16 + rlo)*128 + k0 + 16*hi, 1.f);
          acc = WMMA(a, b, acc);
        }
        return acc;
      };
      v8f ir  = gateacc(Wih, 0, false), hr = gateacc(Whh, 0, true);
      v8f iz  = gateacc(Wih, 1, false), hz = gateacc(Whh, 1, true);
      v8f inn = gateacc(Wih, 2, false), hn = gateacc(Whh, 2, true);
#pragma unroll
      for (int r = 0; r < 8; ++r){
        int col = jt*16 + rlo;
        float rg = sigm(ir[r] + bih[col]       + hr[r] + bhh[col]);
        float zg = sigm(iz[r] + bih[128 + col] + hz[r] + bhh[128 + col]);
        float nn = tanhf(inn[r] + bih[256 + col] + rg*(hn[r] + bhh[256 + col]));
        float ho = hl[rt*16 + r + 8*hi][col];
        hp[i][r] = (1.f - zg)*nn + zg*ho;
      }
    }
    __syncthreads();
#pragma unroll
    for (int i = 0; i < 4; ++i){
      int jt = jb + i;
#pragma unroll
      for (int r = 0; r < 8; ++r)
        hl[rt*16 + r + 8*hi][jt*16 + rlo] = hp[i][r];
    }
    __syncthreads();
  }
  for (int i = tid; i < 64*128; i += 256) hbuf[i] = hl[i>>7][i&127];
}

// ---------------------------------------------------------------- Set2Set LSTM elementwise combine
__global__ void k_lstm(const float* __restrict__ gates,
                       float* __restrict__ cs, float* __restrict__ hs)
{
  int idx = blockIdx.x*blockDim.x + threadIdx.x;
  if (idx >= 64*128) return;
  int g = idx >> 7, j = idx & 127;
  const float* gr = gates + g*512;
  float i_ = sigm(gr[j]);
  float f_ = sigm(gr[128 + j]);
  float g_ = tanhf(gr[256 + j]);
  float o_ = sigm(gr[384 + j]);
  float c  = f_*cs[idx] + i_*g_;
  cs[idx] = c;
  hs[idx] = o_*tanhf(c);
}

// ---------------------------------------------------------------- Set2Set attend: e = zg.hs, softmax, r, q_star
__global__ void __launch_bounds__(128)
k_s2s_attend(const float* __restrict__ zg, const float* __restrict__ hs,
             float* __restrict__ qstar)
{
  int g = blockIdx.x, tid = threadIdx.x;
  __shared__ float ee[64];
  __shared__ float hsh[128];
  if (tid < 128) hsh[tid] = hs[g*128 + tid];
  __syncthreads();
  if (tid < 64){
    float s = 0.f;
    for (int h = 0; h < 128; ++h) s += zg[(g*64 + tid)*128 + h]*hsh[h];
    ee[tid] = s;
  }
  __syncthreads();
  float mx = -1e30f;
  for (int l = 0; l < 64; ++l) mx = fmaxf(mx, ee[l]);
  float sum = 0.f;
  for (int l = 0; l < 64; ++l) sum += __expf(ee[l] - mx);
  float inv = 1.f / sum;
  for (int h = tid; h < 128; h += blockDim.x){
    float r = 0.f;
    for (int l = 0; l < 64; ++l) r += __expf(ee[l] - mx)*inv*zg[(g*64 + l)*128 + h];
    qstar[g*256 + h]       = hsh[h];
    qstar[g*256 + 128 + h] = r;
  }
}

// ---------------------------------------------------------------- attentional aggregation pooling
__global__ void __launch_bounds__(128)
k_pool(const float* __restrict__ score, const float* __restrict__ zg,
       float* __restrict__ out)
{
  int g = blockIdx.x, tid = threadIdx.x;
  __shared__ float ee[64];
  if (tid < 64) ee[tid] = score[g*64 + tid];
  __syncthreads();
  float mx = -1e30f;
  for (int l = 0; l < 64; ++l) mx = fmaxf(mx, ee[l]);
  float sum = 0.f;
  for (int l = 0; l < 64; ++l) sum += __expf(ee[l] - mx);
  float inv = 1.f / sum;
  for (int h = tid; h < 128; h += blockDim.x){
    float r = 0.f;
    for (int l = 0; l < 64; ++l) r += __expf(ee[l] - mx)*inv*zg[(g*64 + l)*128 + h];
    out[g*128 + h] = r;
  }
}

// ---------------------------------------------------------------- row dot: out[r] = X[r,:].w + b
__global__ void __launch_bounds__(256)
k_rowdot(const float* __restrict__ X, const float* __restrict__ wv,
         const float* __restrict__ b, float* __restrict__ out, int rows, int D)
{
  int w = blockIdx.x*8 + (threadIdx.x >> 5);
  if (w >= rows) return;
  int lane = threadIdx.x & 31;
  float s = 0.f;
  for (int c = lane; c < D; c += 32) s += X[w*D + c]*wv[c];
  s = redsum32(s);
  if (lane == 0) out[w] = s + b[0];
}

// ---------------------------------------------------------------- concat [q_star | att_out] -> [64,384]
__global__ void k_concat(const float* __restrict__ qstar,
                         const float* __restrict__ att, float* __restrict__ cat)
{
  int idx = blockIdx.x*blockDim.x + threadIdx.x;
  if (idx >= 64*384) return;
  int g = idx / 384, c = idx % 384;
  cat[idx] = (c < 256) ? qstar[g*256 + c] : att[g*128 + (c - 256)];
}

// ================================================================ host
extern "C" void kernel_launch(void* const* d_in, const int* in_sizes, int n_in,
                              void* d_out, int out_size, void* d_ws, size_t ws_size,
                              hipStream_t stream)
{
  (void)in_sizes; (void)n_in; (void)out_size; (void)ws_size;
  auto in = [&](int i){ return (const float*)d_in[i]; };

  const float* x  = in(0);
  const int*   ei = (const int*)d_in[1];
  const float* ea = in(2);
  // d_in[3] = batch (unused; graphs sorted and equal-sized)

  const float* embW = in(58), *embB = in(59), *embBe = in(60), *embG = in(61);
  const float* gW1 = in(62), *gW2 = in(63), *gb1 = in(64), *gb2 = in(65), *gbe = in(66), *gg = in(67);
  const float* gruWhh = in(68), *gruWih = in(69), *gruBhh = in(70), *gruBih = in(71);
  const float* glnbe = in(72), *glng = in(73);
  const float* r0W = in(74), *r0b = in(75), *r0be = in(76), *r0g = in(77);
  const float* r1W = in(78), *r1b = in(79), *r1be = in(80), *r1g = in(81);
  const float* r2W = in(82), *r2b = in(83);
  const float* s2sWhh = in(84), *s2sWih = in(85), *s2sBhh = in(86), *s2sBih = in(87);

  float* f = (float*)d_ws;
  float* agg    = f;                 f += 4096*128;
  float* z2     = f;                 f += 4096*128;
  float* qkv    = f;                 f += 4096*384;
  float* vtb    = f;                 f += 4096*128;
  float* attno  = f;                 f += 4096*128;
  float* proj   = f;                 f += 4096*128;
  float* zA     = f;                 f += 4096*128;
  float* zB     = f;                 f += 4096*128;
  float* g1b    = f;                 f += 4096*128;
  float* gatev  = f;                 f += 4096;
  float* hbuf   = f;                 f += 64*128;
  float* qstar  = f;                 f += 64*256;
  float* hsb    = f;                 f += 64*128;
  float* csb    = f;                 f += 64*128;
  float* gates  = f;                 f += 64*512;
  float* attout = f;                 f += 64*128;
  float* cat    = f;                 f += 64*384;
  float* t1     = f;                 f += 64*128;
  float* t2     = f;                 f += 64*64;

  auto gemm = [&](const float* A, const float* W, const float* B, float* C,
                  int M, int N, int K){
    int tiles = (M/16)*(N/32);
    k_gemm<false><<<(tiles+3)/4, 128, 0, stream>>>(A, W, B, C, M, N, K);
  };
  auto gemm_acc = [&](const float* A, const float* W, const float* B, float* C,
                      int M, int N, int K){
    int tiles = (M/16)*(N/32);
    k_gemm<true><<<(tiles+3)/4, 128, 0, stream>>>(A, W, B, C, M, N, K);
  };
  auto ln = [&](const float* X, const float* R, const float* g, const float* b,
                const float* base, float* out, int rows, int D, int relu, int shift){
    dim3 grid((rows+7)/8);
    if (R && base)  k_ln<true,true,false><<<grid, 256, 0, stream>>>(X, R, g, b, base, out, rows, D, shift);
    else if (R)     k_ln<true,false,false><<<grid, 256, 0, stream>>>(X, R, g, b, base, out, rows, D, shift);
    else if (relu)  k_ln<false,false,true><<<grid, 256, 0, stream>>>(X, R, g, b, base, out, rows, D, shift);
    else            k_ln<false,false,false><<<grid, 256, 0, stream>>>(X, R, g, b, base, out, rows, D, shift);
  };
  auto zero = [&](float* p, int n){
    k_zero<<<(n+255)/256, 256, 0, stream>>>(p, n);
  };

  // embedding: z = relu(LN(x @ W^T + b))
  gemm(x, embW, embB, z2, 4096, 128, 64);
  ln(z2, nullptr, embG, embBe, nullptr, zA, 4096, 128, 1, 0);
  zero(hbuf, 64*128);                 // GRU hidden carries across layers, init 0

  for (int c = 0; c < 3; ++c){
    int b = 4 + 18*c;
    const float* Wo = in(b+0),  *Wqkv = in(b+1), *bo = in(b+2), *bqkv = in(b+3);
    const float* l1be = in(b+4), *l1g = in(b+5), *l2be = in(b+6), *l2g = in(b+7);
    const float* mW1 = in(b+8),  *mW2 = in(b+9),  *mW3 = in(b+10);
    const float* mb1 = in(b+11), *mb2 = in(b+12), *mb3 = in(b+13);
    const float* mbe1 = in(b+14), *mbe2 = in(b+15), *mg1 = in(b+16), *mg2 = in(b+17);

    zero(agg, 4096*128);
    k_edge<<<1024, 128, 0, stream>>>(zA, ei, ea, mW1, mb1, mg1, mbe1,
                                     mW2, mb2, mg2, mbe2, mW3, mb3, agg);
    ln(zA, agg, l1g, l1be, nullptr, z2, 4096, 128, 0, 0);          // z2 = LN(z+agg)
    gemm(z2, Wqkv, bqkv, qkv, 4096, 384, 128);
    k_vt<<<(128*4096+255)/256, 256, 0, stream>>>(qkv, vtb);
    k_attn<<<dim3(64,4), 128, 0, stream>>>(qkv, vtb, attno);
    gemm(attno, Wo, bo, proj, 4096, 128, 128);
    ln(z2, proj, l2g, l2be, zA, zB, 4096, 128, 0, 0);              // zB = z + LN(z2+mha)
    k_gru<<<1, 256, 0, stream>>>(zB, gruWhh, gruWih, gruBhh, gruBih, hbuf);
    ln(hbuf, nullptr, glng, glnbe, nullptr, zA, 4096, 128, 0, 6);  // zA = LN(bcast h)
  }

  // Set2Set pooling
  zero(qstar, 64*256); zero(hsb, 64*128); zero(csb, 64*128);
  for (int s = 0; s < 6; ++s){
    gemm(qstar, s2sWih, s2sBih, gates, 64, 512, 256);
    gemm_acc(hsb, s2sWhh, s2sBhh, gates, 64, 512, 128);
    k_lstm<<<32, 256, 0, stream>>>(gates, csb, hsb);
    k_s2s_attend<<<64, 128, 0, stream>>>(zA, hsb, qstar);
  }

  // attentional aggregation pooling
  gemm(zA, gW1, gb1, g1b, 4096, 128, 128);
  ln(g1b, nullptr, gg, gbe, nullptr, g1b, 4096, 128, 1, 0);
  k_rowdot<<<512, 256, 0, stream>>>(g1b, gW2, gb2, gatev, 4096, 128);
  k_pool<<<64, 128, 0, stream>>>(gatev, zA, attout);

  // regressor
  k_concat<<<(64*384+255)/256, 256, 0, stream>>>(qstar, attout, cat);
  gemm(cat, r0W, r0b, t1, 64, 128, 384);
  ln(t1, nullptr, r0g, r0be, nullptr, t1, 64, 128, 1, 0);
  gemm(t1, r1W, r1b, t2, 64, 64, 128);
  ln(t2, nullptr, r1g, r1be, nullptr, t2, 64, 64, 1, 0);
  k_rowdot<<<8, 256, 0, stream>>>(t2, r2W, r2b, (float*)d_out, 64, 64);
}
